// MultiHeadedAttentionWithHistoryRelations_35553739276503
// MI455X (gfx1250) — compile-verified, matrix operand order checked by reading, not run on hardware
//
#include <hip/hip_runtime.h>
#include <hip/hip_bf16.h>
#include <math.h>

#define B_ 2
#define S_ 512
#define D_ 1024
#define H_ 8
#define DK_ 128
#define NSEP_ 4
#define TOK_ (B_*S_)

typedef __attribute__((ext_vector_type(16))) __bf16 v16bf;
typedef __attribute__((ext_vector_type(2)))  __bf16 v2bf;
typedef __attribute__((ext_vector_type(8)))  float  v8f;

union Frag { v16bf v; uint4 q[2]; };

#if defined(__has_builtin)
#  if __has_builtin(__builtin_amdgcn_cvt_pk_bf16_f32)
#    define HAVE_CVT_PK_BF16 1
#  endif
#endif

// pack two f32 -> two bf16 in one dword (hardware v_cvt_pk_bf16_f32 when available)
static __device__ __forceinline__ unsigned pk2(float lo, float hi){
#ifdef HAVE_CVT_PK_BF16
  v2bf r = __builtin_amdgcn_cvt_pk_bf16_f32(lo, hi);
  return __builtin_bit_cast(unsigned, r);
#else
  v2bf r = { (__bf16)lo, (__bf16)hi };
  return __builtin_bit_cast(unsigned, r);
#endif
}
static __device__ __forceinline__ __bf16 f2bf(float f){
  unsigned u = pk2(f, 0.f);
  unsigned short h = (unsigned short)(u & 0xFFFFu);
  return __builtin_bit_cast(__bf16, h);
}
static __device__ __forceinline__ float bf2f(__bf16 x){
  unsigned short h = __builtin_bit_cast(unsigned short, x);
  unsigned u = ((unsigned)h) << 16;
  return __builtin_bit_cast(float, u);
}
static __device__ __forceinline__ uint4 pack8f(float4 a, float4 b){
  uint4 r; r.x = pk2(a.x,a.y); r.y = pk2(a.z,a.w); r.z = pk2(b.x,b.y); r.w = pk2(b.z,b.w);
  return r;
}
static __device__ __forceinline__ v8f wmma_bf16(const Frag& a, const Frag& b, v8f c){
  return __builtin_amdgcn_wmma_f32_16x16x32_bf16(false, a.v, false, b.v, (short)0, c, false, false);
}

// ---------------------------------------------------------------- cast f32->bf16
__global__ void cast_kernel(const float* __restrict__ src, __bf16* __restrict__ dst, long n){
  long i = ((long)blockIdx.x * blockDim.x + threadIdx.x) * 8;
  if (i + 8 <= n){
    float4 a = *(const float4*)(src + i);
    float4 b = *(const float4*)(src + i + 4);
    *(uint4*)(dst + i) = pack8f(a, b);
  }
}

// ---------------------------------------------------------------- generic WMMA GEMM: C = A @ W^T + bias
// A: [M,K] bf16 row-major.  W: [N,K] bf16 row-major (so B-matrix columns are W rows: contiguous).
// MODE 0: out bf16 head-split [B,H,S,DK]; MODE 1: out bf16 transposed [B,H,DK,S]; MODE 2: out f32 [M,N]
template<int MODE>
__global__ __launch_bounds__(256) void gemm_bf16_kernel(const __bf16* __restrict__ A,
                                                        const __bf16* __restrict__ W,
                                                        const float*  __restrict__ bias,
                                                        void* __restrict__ out,
                                                        int M, int N, int K){
  const int lane = threadIdx.x & 31, wv = threadIdx.x >> 5;
  const int colb = lane & 15, lh = (lane >> 4) << 3;
  const int m0 = blockIdx.x << 4;
  const int n0 = blockIdx.y * 512 + wv * 64;
  v8f z = {}; v8f acc[4]; for (int t = 0; t < 4; ++t) acc[t] = z;
  const __bf16* Ab = A + (size_t)(m0 + colb) * K;
  for (int kk = 0; kk < K; kk += 32){
    Frag a; a.q[0] = *(const uint4*)(Ab + kk + lh); a.q[1] = *(const uint4*)(Ab + kk + 16 + lh);
#pragma unroll
    for (int t = 0; t < 4; ++t){
      const __bf16* Bb = W + (size_t)(n0 + t*16 + colb) * K + kk;
      Frag b; b.q[0] = *(const uint4*)(Bb + lh); b.q[1] = *(const uint4*)(Bb + 16 + lh);
      acc[t] = wmma_bf16(a, b, acc[t]);
    }
  }
#pragma unroll
  for (int t = 0; t < 4; ++t){
    const int n = n0 + t*16 + colb;
    const float bv = bias[n];
#pragma unroll
    for (int r = 0; r < 8; ++r){
      const int m = m0 + lh + r;
      const float val = acc[t][r] + bv;
      if constexpr (MODE == 2){
        ((float*)out)[(size_t)m * N + n] = val;
      } else {
        const int bb = m >> 9, s = m & (S_-1), h = n >> 7, dk = n & (DK_-1);
        __bf16* o = (__bf16*)out;
        if constexpr (MODE == 0) o[(((size_t)(bb*H_ + h))*S_ + s)*DK_ + dk] = f2bf(val);
        else                     o[(((size_t)(bb*H_ + h))*DK_ + dk)*S_ + s] = f2bf(val);
      }
    }
  }
}

// ---------------------------------------------------------------- sep pooling / gate g / loss
__global__ __launch_bounds__(256) void sep_kernel(const __bf16* __restrict__ qb,
                                                  const float* __restrict__ hw,
                                                  const int* __restrict__ sep_id,
                                                  const float* __restrict__ hreg,
                                                  float* __restrict__ gbuf,
                                                  float* __restrict__ loss_out){
  __shared__ float pred[B_*H_][NSEP_];
  __shared__ float lossp[B_*H_];
  const int tid = threadIdx.x;
  if (tid < B_*H_){
    const int b = tid / H_, h = tid % H_;
    float pool[NSEP_];
#pragma unroll
    for (int n = 0; n < NSEP_; ++n){
      const int s = sep_id[n];
      const __bf16* qr = qb + (((size_t)(b*H_ + h))*S_ + s)*DK_;
      float acc = 0.f;
      for (int d = 0; d < DK_; ++d) acc += bf2f(qr[d]) * hw[h*DK_ + d];
      pool[n] = acc;
    }
    float mx = pool[0];
    for (int n = 1; n < NSEP_; ++n) mx = fmaxf(mx, pool[n]);
    float sum = 0.f;
    for (int n = 0; n < NSEP_; ++n){ pool[n] = __expf(pool[n] - mx); sum += pool[n]; }
    const float inv = 1.f / sum;
    float pmax = 0.f, l2 = 0.f;
    for (int n = 0; n < NSEP_; ++n){
      const float p = pool[n] * inv;
      pred[tid][n] = p;
      pmax = fmaxf(pmax, fabsf(p));
      const float t = hreg[n];
      l2 += t * (logf(t) - logf(p));
    }
    lossp[tid] = (1.f - pmax) / (float)(B_*H_) + l2 / (float)(B_*H_*NSEP_);
  }
  __syncthreads();
  if (tid == 0){
    float L = 0.f;
    for (int i = 0; i < B_*H_; ++i) L += lossp[i];
    loss_out[0] = L;
  }
  for (int idx = tid; idx < B_*H_*S_; idx += blockDim.x){
    const int p = idx & (S_-1);
    const int bh = idx >> 9;
    float gv = 0.f;
#pragma unroll
    for (int n = 0; n < NSEP_; ++n) gv += pred[bh][n] * ((p >= sep_id[n]) ? 1.f : 0.f);
    gbuf[idx] = gv;
  }
}

// ---------------------------------------------------------------- rel-K scores: scores[b,h,q,k] = q[b,h,q,:] . relation_k[b,q,k,:]
// one block per (b,q); A = 8 head-rows padded to 16; streams relation_k f32 -> bf16 via v_cvt_pk_bf16_f32
__global__ __launch_bounds__(256) void relscores_kernel(const __bf16* __restrict__ qb,
                                                        const float* __restrict__ relk,
                                                        float* __restrict__ scores){
  const int b = blockIdx.x / S_, q = blockIdx.x % S_;
  const int lane = threadIdx.x & 31, wv = threadIdx.x >> 5;
  const int colb = lane & 15, lh = (lane >> 4) << 3;
  const int m = colb;                       // head row (pad rows 8..15)
  v8f z = {}; v8f acc[4]; for (int t = 0; t < 4; ++t) acc[t] = z;
  for (int kk = 0; kk < DK_; kk += 32){
    Frag a;
    if (m < H_){
      const __bf16* Ab = qb + (((size_t)(b*H_ + m))*S_ + q)*DK_ + kk;
      a.q[0] = *(const uint4*)(Ab + lh); a.q[1] = *(const uint4*)(Ab + 16 + lh);
    } else { a.q[0] = make_uint4(0,0,0,0); a.q[1] = make_uint4(0,0,0,0); }
#pragma unroll
    for (int t = 0; t < 4; ++t){
      const int kn = (wv*4 + t)*16 + colb;
      const float* Bb = relk + (((size_t)(b*S_ + q)*S_ + kn)*DK_ + kk);
      __builtin_prefetch(Bb + DK_, 0, 1);   // global_prefetch of next row stripe
      float4 f0 = *(const float4*)(Bb + lh);
      float4 f1 = *(const float4*)(Bb + lh + 4);
      float4 f2 = *(const float4*)(Bb + 16 + lh);
      float4 f3 = *(const float4*)(Bb + 20 + lh);
      Frag bb; bb.q[0] = pack8f(f0, f1); bb.q[1] = pack8f(f2, f3);
      acc[t] = wmma_bf16(a, bb, acc[t]);
    }
  }
  if (lane < 16){                            // rows 0..7 are real heads
#pragma unroll
    for (int t = 0; t < 4; ++t){
      const int kn = (wv*4 + t)*16 + colb;
#pragma unroll
      for (int r = 0; r < 8; ++r)
        scores[(((size_t)(b*H_ + r))*S_ + q)*S_ + kn] = acc[t][r];
    }
  }
}

// ---------------------------------------------------------------- main scores: += q@k^T, then scale/mask/gate
__global__ __launch_bounds__(256) void scores_kernel(const __bf16* __restrict__ qb,
                                                     const __bf16* __restrict__ kb,
                                                     const float* __restrict__ gbuf,
                                                     const int* __restrict__ mask,
                                                     float* __restrict__ scores){
  const int bh = blockIdx.x, b = bh / H_;
  const int q0 = blockIdx.y << 4;
  const int lane = threadIdx.x & 31, wv = threadIdx.x >> 5;
  const int colb = lane & 15, lh = (lane >> 4) << 3;
  v8f z = {}; v8f acc[4]; for (int t = 0; t < 4; ++t) acc[t] = z;
  const __bf16* Ab = qb + ((size_t)bh*S_ + q0 + colb)*DK_;
  for (int kk = 0; kk < DK_; kk += 32){
    Frag a; a.q[0] = *(const uint4*)(Ab + kk + lh); a.q[1] = *(const uint4*)(Ab + kk + 16 + lh);
#pragma unroll
    for (int t = 0; t < 4; ++t){
      const int n = (wv*4 + t)*16 + colb;
      const __bf16* Bb = kb + ((size_t)bh*S_ + n)*DK_ + kk;
      Frag bb; bb.q[0] = *(const uint4*)(Bb + lh); bb.q[1] = *(const uint4*)(Bb + 16 + lh);
      acc[t] = wmma_bf16(a, bb, acc[t]);
    }
  }
  const float scale = 0.0883883476483184f;   // 1/sqrt(128)
#pragma unroll
  for (int t = 0; t < 4; ++t){
    const int n = (wv*4 + t)*16 + colb;
#pragma unroll
    for (int r = 0; r < 8; ++r){
      const int mrow = q0 + lh + r;
      const size_t idx = ((size_t)bh*S_ + mrow)*S_ + n;
      float val = (scores[idx] + acc[t][r]) * scale;
      if (mask[((size_t)b*S_ + mrow)*S_ + n] == 0) val = -1000000000.0f;
      const int mn = mrow < n ? mrow : n;
      val *= gbuf[(size_t)bh*S_ + mn];
      scores[idx] = val;
    }
  }
}

// ---------------------------------------------------------------- softmax (wave per row, wave32 shuffles) -> bf16 P
__global__ __launch_bounds__(256) void softmax_kernel(const float* __restrict__ scores,
                                                      __bf16* __restrict__ pb){
  const int lane = threadIdx.x & 31, wv = threadIdx.x >> 5;
  const size_t row = (size_t)blockIdx.x * 8 + wv;
  const float* sr = scores + row * S_;
  float v[16];
  const float4* s4 = (const float4*)(sr + lane*16);
#pragma unroll
  for (int j = 0; j < 4; ++j){
    float4 f = s4[j];
    v[j*4+0]=f.x; v[j*4+1]=f.y; v[j*4+2]=f.z; v[j*4+3]=f.w;
  }
  float mx = -3.0e38f;
#pragma unroll
  for (int i = 0; i < 16; ++i) mx = fmaxf(mx, v[i]);
#pragma unroll
  for (int off = 16; off >= 1; off >>= 1) mx = fmaxf(mx, __shfl_xor(mx, off, 32));
  float sum = 0.f;
#pragma unroll
  for (int i = 0; i < 16; ++i){ v[i] = __expf(v[i] - mx); sum += v[i]; }
#pragma unroll
  for (int off = 16; off >= 1; off >>= 1) sum += __shfl_xor(sum, off, 32);
  const float inv = 1.f / sum;
  uint4* d4 = (uint4*)(pb + row*S_ + lane*16);
#pragma unroll
  for (int j = 0; j < 2; ++j){
    float4 a = make_float4(v[j*8+0]*inv, v[j*8+1]*inv, v[j*8+2]*inv, v[j*8+3]*inv);
    float4 b = make_float4(v[j*8+4]*inv, v[j*8+5]*inv, v[j*8+6]*inv, v[j*8+7]*inv);
    d4[j] = pack8f(a, b);
  }
}

// ---------------------------------------------------------------- X = P @ V  (V pre-transposed -> contiguous B columns)
__global__ __launch_bounds__(256) void pv_kernel(const __bf16* __restrict__ pb,
                                                 const __bf16* __restrict__ vtb,
                                                 float* __restrict__ xf){
  const int bh = blockIdx.x, b = bh / H_, h = bh % H_;
  const int q0 = blockIdx.y << 4;
  const int lane = threadIdx.x & 31, wv = threadIdx.x >> 5;
  const int colb = lane & 15, lh = (lane >> 4) << 3;
  const int n = wv*16 + colb;               // dk column
  v8f acc = {};
  const __bf16* Ab = pb + ((size_t)bh*S_ + q0 + colb)*S_;
  const __bf16* Bb = vtb + ((size_t)bh*DK_ + n)*S_;
  for (int kk = 0; kk < S_; kk += 32){
    Frag a; a.q[0] = *(const uint4*)(Ab + kk + lh); a.q[1] = *(const uint4*)(Ab + kk + 16 + lh);
    Frag bb; bb.q[0] = *(const uint4*)(Bb + kk + lh); bb.q[1] = *(const uint4*)(Bb + kk + 16 + lh);
    acc = wmma_bf16(a, bb, acc);
  }
#pragma unroll
  for (int r = 0; r < 8; ++r){
    const int m = q0 + lh + r;
    xf[((size_t)b*S_ + m)*D_ + h*DK_ + n] = acc[r];
  }
}

// ---------------------------------------------------------------- X += P @ relation_v (LDS-transposed staging, paired b32 stores)
__global__ __launch_bounds__(256) void prelv_kernel(const __bf16* __restrict__ pb,
                                                    const float* __restrict__ relv,
                                                    float* __restrict__ xf){
  __shared__ __bf16 ldsT[DK_ * 40];         // [d][k] transposed tile, stride 40 (16B-aligned rows)
  const int b = blockIdx.x / S_, q = blockIdx.x % S_;
  const int tid = threadIdx.x;
  const int lane = tid & 31, wv = tid >> 5;
  const int colb = lane & 15, lh = (lane >> 4) << 3;
  const int m = colb;                       // head row (pad 8..15)
  const int nd = wv*16 + colb;              // dk column (8 waves cover 128)
  const int kloc2 = (tid >> 4) * 2;         // 0,2,...,30 : pair of k-rows
  const int dbase = (tid & 15) * 8;         // 0..120 : 8 d values
  v8f acc = {};
  for (int kk = 0; kk < S_; kk += 32){
    // cooperative load 32x128 f32 tile of relation_v[b,q]; pack (k,k+1) bf16 pairs -> ds_store_b32 transposed
    {
      const float* src0 = relv + (((size_t)(b*S_ + q)*S_ + kk + kloc2)*DK_ + dbase);
      const float* src1 = src0 + DK_;
      float4 a0 = *(const float4*)(src0), a1 = *(const float4*)(src0 + 4);
      float4 b0 = *(const float4*)(src1), b1 = *(const float4*)(src1 + 4);
      float r0[8] = {a0.x,a0.y,a0.z,a0.w,a1.x,a1.y,a1.z,a1.w};
      float r1[8] = {b0.x,b0.y,b0.z,b0.w,b1.x,b1.y,b1.z,b1.w};
#pragma unroll
      for (int j = 0; j < 8; ++j)
        *(unsigned*)&ldsT[(size_t)(dbase + j)*40 + kloc2] = pk2(r0[j], r1[j]);
    }
    __syncthreads();
    Frag a;
    if (m < H_){
      const __bf16* Ab = pb + (((size_t)(b*H_ + m))*S_ + q)*S_ + kk;
      a.q[0] = *(const uint4*)(Ab + lh); a.q[1] = *(const uint4*)(Ab + 16 + lh);
    } else { a.q[0] = make_uint4(0,0,0,0); a.q[1] = make_uint4(0,0,0,0); }
    const __bf16* Bb = &ldsT[nd * 40];
    Frag bb; bb.q[0] = *(const uint4*)(Bb + lh); bb.q[1] = *(const uint4*)(Bb + 16 + lh);
    acc = wmma_bf16(a, bb, acc);
    __syncthreads();
  }
  if (lane < 16){
#pragma unroll
    for (int r = 0; r < 8; ++r)
      xf[((size_t)b*S_ + q)*D_ + r*DK_ + nd] += acc[r];
  }
}

// ---------------------------------------------------------------- host-side orchestration
extern "C" void kernel_launch(void* const* d_in, const int* in_sizes, int n_in,
                              void* d_out, int out_size, void* d_ws, size_t ws_size,
                              hipStream_t stream){
  (void)in_sizes; (void)n_in; (void)out_size; (void)ws_size;
  const float* query = (const float*)d_in[0];
  const float* key_  = (const float*)d_in[1];
  const float* value = (const float*)d_in[2];
  const float* relk  = (const float*)d_in[3];
  const float* relv  = (const float*)d_in[4];
  const int*   mask  = (const int*)d_in[5];
  const int*   sep   = (const int*)d_in[6];
  const float* hreg  = (const float*)d_in[7];
  const float* Wq = (const float*)d_in[8];  const float* bq = (const float*)d_in[9];
  const float* Wk = (const float*)d_in[10]; const float* bk = (const float*)d_in[11];
  const float* Wv = (const float*)d_in[12]; const float* bv = (const float*)d_in[13];
  const float* Wo = (const float*)d_in[14]; const float* bo = (const float*)d_in[15];
  const float* hw = (const float*)d_in[16];
  float* out = (float*)d_out;

  char* ws = (char*)d_ws;
  size_t off = 0;
  auto alloc = [&](size_t bytes)->void*{
    off = (off + 255) & ~(size_t)255;
    void* p = ws + off; off += bytes; return p;
  };
  const size_t NT = (size_t)TOK_ * D_;      // 1M elements
  __bf16* xq = (__bf16*)alloc(NT*2);
  __bf16* xk = (__bf16*)alloc(NT*2);
  __bf16* xv = (__bf16*)alloc(NT*2);
  __bf16* wqb = (__bf16*)alloc(NT*2);
  __bf16* wkb = (__bf16*)alloc(NT*2);
  __bf16* wvb = (__bf16*)alloc(NT*2);
  __bf16* wob = (__bf16*)alloc(NT*2);
  __bf16* qb  = (__bf16*)alloc((size_t)B_*H_*S_*DK_*2);
  __bf16* kb  = (__bf16*)alloc((size_t)B_*H_*S_*DK_*2);
  __bf16* vtb = (__bf16*)alloc((size_t)B_*H_*DK_*S_*2);
  float*  scores = (float*)alloc((size_t)B_*H_*S_*S_*4);
  __bf16* pbuf   = (__bf16*)alloc((size_t)B_*H_*S_*S_*2);
  float*  gbuf   = (float*)alloc((size_t)B_*H_*S_*4);
  float*  xf     = (float*)alloc((size_t)B_*S_*D_*4);
  __bf16* xb     = (__bf16*)alloc((size_t)B_*S_*D_*2);

  const dim3 blk(256);
  const int castGrid = (int)(NT/8/256);
  // casts
  cast_kernel<<<castGrid, blk, 0, stream>>>(query, xq, (long)NT);
  cast_kernel<<<castGrid, blk, 0, stream>>>(key_,  xk, (long)NT);
  cast_kernel<<<castGrid, blk, 0, stream>>>(value, xv, (long)NT);
  cast_kernel<<<castGrid, blk, 0, stream>>>(Wq, wqb, (long)NT);
  cast_kernel<<<castGrid, blk, 0, stream>>>(Wk, wkb, (long)NT);
  cast_kernel<<<castGrid, blk, 0, stream>>>(Wv, wvb, (long)NT);
  cast_kernel<<<castGrid, blk, 0, stream>>>(Wo, wob, (long)NT);
  // projections (WMMA)
  const dim3 gg(TOK_/16, D_/512);
  gemm_bf16_kernel<0><<<gg, blk, 0, stream>>>(xq, wqb, bq, qb,  TOK_, D_, D_);
  gemm_bf16_kernel<0><<<gg, blk, 0, stream>>>(xk, wkb, bk, kb,  TOK_, D_, D_);
  gemm_bf16_kernel<1><<<gg, blk, 0, stream>>>(xv, wvb, bv, vtb, TOK_, D_, D_);
  // sep pooling / gate / loss (writes loss scalar at d_out[B*S*D])
  sep_kernel<<<1, blk, 0, stream>>>(qb, hw, sep, hreg, gbuf, out + (size_t)B_*S_*D_);
  // relation-K scores (writes scores), then main q@k^T + scale/mask/gate
  relscores_kernel<<<B_*S_, blk, 0, stream>>>(qb, relk, scores);
  scores_kernel<<<dim3(B_*H_, S_/16), blk, 0, stream>>>(qb, kb, gbuf, mask, scores);
  // softmax -> bf16 P
  softmax_kernel<<<(B_*H_*S_)/8, blk, 0, stream>>>(scores, pbuf);
  // X = P@V ; X += P@relation_v
  pv_kernel<<<dim3(B_*H_, S_/16), blk, 0, stream>>>(pbuf, vtb, xf);
  prelv_kernel<<<B_*S_, blk, 0, stream>>>(pbuf, relv, xf);
  // output projection
  cast_kernel<<<castGrid, blk, 0, stream>>>(xf, xb, (long)NT);
  gemm_bf16_kernel<2><<<gg, blk, 0, stream>>>(xb, wob, bo, out, TOK_, D_, D_);
}
